// CrossAttnBlock_60447369724149
// MI455X (gfx1250) — compile-verified, matrix operand order checked by reading, not run on hardware
//
#include <hip/hip_runtime.h>
#include <cstdint>

// Problem constants (from reference): B, LQ, E, H, D, KVD = 2, 8192, 768, 12, 64, 4096
#define Bn    2
#define LQn   8192
#define En    768
#define Hn    12
#define Dh    64
#define KVDn  4096
#define NTOK  512      // total KV tokens (sum LENS)
#define TWOE  1536
#define LDTK  72       // padded LDS row stride (halfs) for BK=64 GEMM tiles (144B, 16B-aligned)

typedef __attribute__((ext_vector_type(16))) __bf16 v16bf;
typedef __attribute__((ext_vector_type(8)))  __bf16 bf16x8;
typedef __attribute__((ext_vector_type(8)))  float  v8f;

__device__ __forceinline__ v8f wmma_bf16(v16bf a, v16bf b, v8f c) {
  // D = A(16x32 bf16) x B(32x16 bf16) + C(16x16 f32)
  return __builtin_amdgcn_wmma_f32_16x16x32_bf16(false, a, false, b, (short)0, c, false, false);
}

// Load a 16x32 bf16 fragment from a row-major matrix (row = M for A, row = N for B of A*B^T).
// CDNA5 layout: lanes 0-15 hold rows 0-15 with K {0..7,16..23}; lanes 16-31 hold K {8..15,24..31}.
__device__ __forceinline__ v16bf load_frag(const __bf16* base, int ld, int lane) {
  const int r  = lane & 15;
  const int kb = (lane >> 4) << 3;            // 0 or 8
  bf16x8 lo = *(const bf16x8*)(base + (size_t)r * ld + kb);
  bf16x8 hi = *(const bf16x8*)(base + (size_t)r * ld + kb + 16);
  return __builtin_shufflevector(lo, hi, 0,1,2,3,4,5,6,7,8,9,10,11,12,13,14,15);
}

__device__ __forceinline__ bf16x8 cvt8(float4 a, float4 b) {
  bf16x8 p = { (__bf16)a.x,(__bf16)a.y,(__bf16)a.z,(__bf16)a.w,
               (__bf16)b.x,(__bf16)b.y,(__bf16)b.z,(__bf16)b.w };
  return p;
}

// ---------------------------------------------------------------------------
// C = A(MxK) * B(NxK)^T (+bias[col]) with on-the-fly f32->bf16 conversion.
// Block: 256 threads = 8 waves; BM=128 (16-row strip per wave), BN=32 (2 tiles
// per wave sharing one A fragment), BK=64, double-buffered LDS +
// distance-2 global_prefetch so staged b128 loads hit GL2/WGP$.
// MODE 0: write f32 row-major [M][N].  MODE 1: write bf16 qp head layout [b][h][q][d].
// ---------------------------------------------------------------------------
template<int ABF16, int MODE>
__global__ __launch_bounds__(256) void gemm_wmma_abT(
    const void* __restrict__ Ap, const float* __restrict__ Bw,
    const float* __restrict__ bias, void* __restrict__ outp,
    int M, int N, int K)
{
  __shared__ __bf16 As[2][128 * LDTK];
  __shared__ __bf16 Bs[2][32 * LDTK];
  const int tid  = threadIdx.x;
  const int lane = tid & 31;
  const int w    = tid >> 5;
  const int col0 = blockIdx.x * 32;
  const int row0 = blockIdx.y * 128;

  // staging coordinates (compile-time shaped)
  const int arr = tid >> 1;              // A row 0..127
  const int acc_ = (tid & 1) * 32;       // A col half: 0 or 32
  const int brr = tid >> 3;              // B row 0..31
  const int bcc = (tid & 7) * 8;         // B col: 0..56 step 8

  const __bf16* Ab = (const __bf16*)Ap;
  const float*  Af = (const float*)Ap;

  auto stage = [&](int buf, int k0) {
    __bf16* as = &As[buf][0];
    __bf16* bs = &Bs[buf][0];
    if (ABF16) {
      const bf16x8* src = (const bf16x8*)(Ab + (size_t)(row0 + arr) * K + k0 + acc_);
      #pragma unroll
      for (int j = 0; j < 4; ++j)
        *(bf16x8*)&as[arr * LDTK + acc_ + 8 * j] = src[j];
    } else {
      const float4* src = (const float4*)(Af + (size_t)(row0 + arr) * K + k0 + acc_);
      float4 x0 = src[0], x1 = src[1], x2 = src[2], x3 = src[3];
      float4 x4 = src[4], x5 = src[5], x6 = src[6], x7 = src[7];
      *(bf16x8*)&as[arr * LDTK + acc_ +  0] = cvt8(x0, x1);
      *(bf16x8*)&as[arr * LDTK + acc_ +  8] = cvt8(x2, x3);
      *(bf16x8*)&as[arr * LDTK + acc_ + 16] = cvt8(x4, x5);
      *(bf16x8*)&as[arr * LDTK + acc_ + 24] = cvt8(x6, x7);
    }
    const float4* bsrc = (const float4*)(Bw + (size_t)(col0 + brr) * K + k0 + bcc);
    float4 y0 = bsrc[0], y1 = bsrc[1];
    *(bf16x8*)&bs[brr * LDTK + bcc] = cvt8(y0, y1);
  };

  auto prefetch = [&](int k0) {          // pull chunk k0 toward the WGP caches
    if (ABF16) {
      __builtin_prefetch(Ab + (size_t)(row0 + arr) * K + k0 + acc_, 0, 3);
    } else {
      __builtin_prefetch(Af + (size_t)(row0 + arr) * K + k0 + acc_, 0, 3);
    }
    __builtin_prefetch(Bw + (size_t)(col0 + brr) * K + k0 + bcc, 0, 3);
  };

  v8f acc0 = {}, acc1 = {};
  const int nk = K >> 6;                 // K is a multiple of 64 for all uses
  stage(0, 0);
  if (nk > 1) prefetch(64);
  __syncthreads();
  for (int t = 0; t < nk; ++t) {
    const int cur = t & 1;
    if (t + 1 < nk) stage(cur ^ 1, (t + 1) << 6);   // fill other buffer under compute
    if (t + 2 < nk) prefetch((t + 2) << 6);         // distance-2 cache prefetch
    const __bf16* as = &As[cur][w * 16 * LDTK];
    const __bf16* bs = &Bs[cur][0];
    #pragma unroll
    for (int kk = 0; kk < 64; kk += 32) {
      v16bf a  = load_frag(as + kk, LDTK, lane);
      v16bf b0 = load_frag(bs + kk, LDTK, lane);
      v16bf b1 = load_frag(bs + 16 * LDTK + kk, LDTK, lane);
      acc0 = wmma_bf16(a, b0, acc0);
      acc1 = wmma_bf16(a, b1, acc1);
    }
    __syncthreads();
  }

  // epilogue: C layout = VGPR r -> row (lane<16 ? r : 8+r), col = lane&15
  const int nloc = lane & 15;
  const int mh   = (lane >> 4) << 3;
  #pragma unroll
  for (int c = 0; c < 2; ++c) {
    const int col = col0 + c * 16 + nloc;
    const float bv = bias ? bias[col] : 0.0f;
    const v8f acc = c ? acc1 : acc0;
    #pragma unroll
    for (int r = 0; r < 8; ++r) {
      const int row = row0 + w * 16 + mh + r;
      const float val = acc[r] + bv;
      if (MODE == 0) {
        ((float*)outp)[(size_t)row * N + col] = val;
      } else {
        const int bb = row >> 13, qq = row & (LQn - 1);
        const int hh = col >> 6,  dd = col & 63;
        ((__bf16*)outp)[(((size_t)bb * Hn + hh) * LQn + qq) * Dh + dd] = (__bf16)val;
      }
    }
  }
}

// ---------------------------------------------------------------------------
// Scatter kv projection: add v_bias to V half, emit K as bf16 [h][n][d] and
// V transposed as bf16 [h][d][n] (so attention B-fragments are row-major gathers).
// ---------------------------------------------------------------------------
__global__ __launch_bounds__(256) void kv_scatter(
    const float* __restrict__ kvraw, const float* __restrict__ v_bias,
    __bf16* __restrict__ kbf, __bf16* __restrict__ vtbf)
{
  const int idx = blockIdx.x * blockDim.x + threadIdx.x;
  if (idx >= NTOK * TWOE) return;
  const int n = idx / TWOE, e2 = idx - n * TWOE;
  float val = kvraw[idx];
  if (e2 < En) {
    const int h = e2 >> 6, d = e2 & 63;
    kbf[((size_t)h * NTOK + n) * Dh + d] = (__bf16)val;
  } else {
    const int e = e2 - En;
    val += v_bias[e];
    const int h = e >> 6, d = e & 63;
    vtbf[((size_t)h * Dh + d) * NTOK + n] = (__bf16)val;
  }
}

// ---------------------------------------------------------------------------
// Attention: one wave per (b, h, 16-query tile). QK^T via WMMA into LDS logits,
// segment softmax, probs(bf16) * V via WMMA -> bf16 ctx [B*LQ][E].
// Segment restriction == the -1e30 mask (out-of-segment probs are exactly 0).
// ---------------------------------------------------------------------------
__global__ __launch_bounds__(32) void attn_wmma(
    const __bf16* __restrict__ qp, const __bf16* __restrict__ kbf,
    const __bf16* __restrict__ vtbf, const int* __restrict__ cu,
    __bf16* __restrict__ ctx)
{
  __shared__ float  sl[16 * 256];
  __shared__ __bf16 sp[16 * 256];
  const int lane = threadIdx.x;
  const int qt = blockIdx.x, h = blockIdx.y, b = blockIdx.z;
  const int lo  = cu[b];
  const int len = cu[b + 1] - lo;          // 256 here; multiple of 32, <= 256

  const __bf16* qbase = qp + (((size_t)b * Hn + h) * LQn + (size_t)qt * 16) * Dh;
  const v16bf qa0 = load_frag(qbase +  0, Dh, lane);
  const v16bf qa1 = load_frag(qbase + 32, Dh, lane);

  const int nloc = lane & 15;
  const int mh   = (lane >> 4) << 3;

  // logits = scale * Q K^T, segment only
  for (int nt = 0; nt < (len >> 4); ++nt) {
    const __bf16* kbase = kbf + ((size_t)h * NTOK + lo + nt * 16) * Dh;
    v8f acc = {};
    acc = wmma_bf16(qa0, load_frag(kbase,      Dh, lane), acc);
    acc = wmma_bf16(qa1, load_frag(kbase + 32, Dh, lane), acc);
    #pragma unroll
    for (int r = 0; r < 8; ++r)
      sl[(mh + r) * 256 + nt * 16 + nloc] = acc[r] * 0.125f;   // 1/sqrt(64)
  }
  __syncthreads();

  // softmax: 2 lanes per row, each covers half the columns, combine via shfl_xor(16)
  const int row  = lane & 15;
  const int c0   = (lane >> 4) * (len >> 1);
  const int c1   = c0 + (len >> 1);
  float mx = -3.0e38f;
  for (int c = c0; c < c1; ++c) mx = fmaxf(mx, sl[row * 256 + c]);
  mx = fmaxf(mx, __shfl_xor(mx, 16));
  float sum = 0.f;
  for (int c = c0; c < c1; ++c) sum += __expf(sl[row * 256 + c] - mx);
  sum += __shfl_xor(sum, 16);
  const float inv = 1.0f / sum;
  for (int c = c0; c < c1; ++c)
    sp[row * 256 + c] = (__bf16)(__expf(sl[row * 256 + c] - mx) * inv);
  __syncthreads();

  // out(16x64) = probs(16xlen) * V(lenx64), V held transposed [d][n]
  v8f o0 = {}, o1 = {}, o2 = {}, o3 = {};
  for (int nt = 0; nt < (len >> 5); ++nt) {
    const v16bf pa = load_frag(&sp[nt * 32], 256, lane);
    const __bf16* vb = vtbf + (size_t)h * Dh * NTOK + lo + nt * 32;
    o0 = wmma_bf16(pa, load_frag(vb,                      NTOK, lane), o0);
    o1 = wmma_bf16(pa, load_frag(vb + (size_t)16 * NTOK,  NTOK, lane), o1);
    o2 = wmma_bf16(pa, load_frag(vb + (size_t)32 * NTOK,  NTOK, lane), o2);
    o3 = wmma_bf16(pa, load_frag(vb + (size_t)48 * NTOK,  NTOK, lane), o3);
  }

  const size_t rowbase = (size_t)b * LQn + (size_t)qt * 16;
  #pragma unroll
  for (int r = 0; r < 8; ++r) {
    const size_t rw = (rowbase + mh + r) * En + (size_t)h * Dh;
    ctx[rw +  0 + nloc] = (__bf16)o0[r];
    ctx[rw + 16 + nloc] = (__bf16)o1[r];
    ctx[rw + 32 + nloc] = (__bf16)o2[r];
    ctx[rw + 48 + nloc] = (__bf16)o3[r];
  }
}

// ---------------------------------------------------------------------------
// Probe importance: imp[b,q] = mean_h ( 1 / sum_p exp(l_p - max_p l) )
// key_t[b]: b==0 -> K rows, b==1 -> V rows (B==2 aliases the kv selector axis).
// Online max/sum-exp, no logit array needed.
// ---------------------------------------------------------------------------
__global__ __launch_bounds__(256) void probe_imp(
    const __bf16* __restrict__ qp, const __bf16* __restrict__ kbf,
    const __bf16* __restrict__ vtbf, const int* __restrict__ probe_ids,
    int P, float* __restrict__ imp)
{
  const int gid = blockIdx.x * blockDim.x + threadIdx.x;
  if (gid >= Bn * LQn) return;
  const int b = gid >> 13;
  const int q = gid & (LQn - 1);
  float acc = 0.f;
  for (int h = 0; h < Hn; ++h) {
    const __bf16* qv = qp + (((size_t)b * Hn + h) * LQn + q) * Dh;
    float qf[Dh];
    #pragma unroll
    for (int d = 0; d < Dh; ++d) qf[d] = (float)qv[d];
    float m = -3.0e38f, s = 0.f;
    for (int p = 0; p < P; ++p) {
      const int pid = probe_ids[p];
      float dot = 0.f;
      if (b == 0) {
        const __bf16* kv = kbf + ((size_t)h * NTOK + pid) * Dh;
        #pragma unroll
        for (int d = 0; d < Dh; ++d) dot += qf[d] * (float)kv[d];
      } else {
        const __bf16* vv = vtbf + (size_t)h * Dh * NTOK + pid;
        #pragma unroll
        for (int d = 0; d < Dh; ++d) dot += qf[d] * (float)vv[(size_t)d * NTOK];
      }
      if (dot > m) { s = s * __expf(m - dot) + 1.f; m = dot; }
      else         { s += __expf(dot - m); }
    }
    acc += 1.f / s;
  }
  imp[gid] = acc * (1.0f / Hn);
}

// ---------------------------------------------------------------------------
extern "C" void kernel_launch(void* const* d_in, const int* in_sizes, int n_in,
                              void* d_out, int out_size, void* d_ws, size_t ws_size,
                              hipStream_t stream)
{
  (void)n_in; (void)out_size; (void)ws_size;
  const float* q_in   = (const float*)d_in[0];
  const float* kvc    = (const float*)d_in[1];
  const float* W_q    = (const float*)d_in[2];
  const float* b_q    = (const float*)d_in[3];
  const float* W_kv   = (const float*)d_in[4];
  const float* v_bias = (const float*)d_in[5];
  const float* W_proj = (const float*)d_in[6];
  const float* b_proj = (const float*)d_in[7];
  const int*   cu     = (const int*)d_in[8];
  const int*   pids   = (const int*)d_in[10];
  const int P = in_sizes[10];
  float* out = (float*)d_out;

  // workspace carve-up (all slices 16B-aligned): ~55 MB total
  char* ws = (char*)d_ws;
  float*  kvraw = (float*)ws;   ws += (size_t)NTOK * TWOE * sizeof(float);          // 3.0 MB
  __bf16* kbf   = (__bf16*)ws;  ws += (size_t)Hn * NTOK * Dh * sizeof(__bf16);      // 0.75 MB
  __bf16* vtbf  = (__bf16*)ws;  ws += (size_t)Hn * Dh * NTOK * sizeof(__bf16);      // 0.75 MB
  __bf16* qpbf  = (__bf16*)ws;  ws += (size_t)Bn * Hn * LQn * Dh * sizeof(__bf16);  // 25.2 MB
  __bf16* ctxbf = (__bf16*)ws;  ws += (size_t)Bn * LQn * En * sizeof(__bf16);       // 25.2 MB

  // 1) kv = kv_compact @ W_kv^T             (512 x 4096 x 1536)
  gemm_wmma_abT<0, 0><<<dim3(TWOE / 32, NTOK / 128), 256, 0, stream>>>(
      kvc, W_kv, nullptr, kvraw, NTOK, TWOE, KVDn);
  // 2) split K / V(+bias), bf16, V transposed
  kv_scatter<<<(NTOK * TWOE + 255) / 256, 256, 0, stream>>>(kvraw, v_bias, kbf, vtbf);
  // 3) qp = q @ W_q^T + b_q  -> bf16 [b][h][q][d]   (16384 x 768 x 768)
  gemm_wmma_abT<0, 1><<<dim3(En / 32, (Bn * LQn) / 128), 256, 0, stream>>>(
      q_in, W_q, b_q, qpbf, Bn * LQn, En, En);
  // 4) attention -> ctx bf16 [B*LQ][E]
  attn_wmma<<<dim3(LQn / 16, Hn, Bn), 32, 0, stream>>>(qpbf, kbf, vtbf, cu, ctxbf);
  // 5) out = ctx @ W_proj^T + b_proj -> d_out f32   (16384 x 768 x 768)
  gemm_wmma_abT<1, 0><<<dim3(En / 32, (Bn * LQn) / 128), 256, 0, stream>>>(
      ctxbf, W_proj, b_proj, out, Bn * LQn, En, En);
  // 6) probe importance -> d_out tail
  probe_imp<<<((Bn * LQn) + 255) / 256, 256, 0, stream>>>(
      qpbf, kbf, vtbf, pids, P, out + (size_t)Bn * LQn * En);
}
